// Net_PPF_LRBF2_84954453115110
// MI455X (gfx1250) — matrix operand order repaired and checked by reading, not compiled, not run on hardware
//
#include <hip/hip_runtime.h>
#include <hip/hip_bf16.h>

typedef __attribute__((ext_vector_type(16))) _Float16 v16h;
typedef __attribute__((ext_vector_type(2)))  __fp16   fp16x2;
typedef __attribute__((ext_vector_type(8)))  float    v8f;

#define LN_EPS 1e-5f

// ---------------------------------------------------------------- utilities

__device__ inline float wave_sum(float v) {
#pragma unroll
  for (int o = 16; o > 0; o >>= 1) v += __shfl_xor(v, o, 32);
  return v;
}

// A fragment: 16x32 f16 tile from row-major fp32 A[r*lda + k].
// ISA 16-bit A layout: lane holds row M=lane&15; kh=lane>>4 selects the K
// sub-block. Element j covers k = kb + (j>>3)*16 + kh*8 + (j&7): two runs of
// 8 consecutive floats -> 4x float4 loads. Caller clamps r in-bounds; rows
// >= nrows produce unused (unstored) D rows, so no zeroing is needed.
__device__ inline v16h load_a_frag(const float* __restrict__ A, int r, int lda,
                                   int kb, int kh) {
  const float* p = A + (size_t)r * lda + kb + (kh << 3);
  float4 x0 = *(const float4*)(p + 0);
  float4 x1 = *(const float4*)(p + 4);
  float4 x2 = *(const float4*)(p + 16);
  float4 x3 = *(const float4*)(p + 20);
  union { v16h v; fp16x2 h[8]; } u;
  u.h[0] = __builtin_amdgcn_cvt_pkrtz(x0.x, x0.y);
  u.h[1] = __builtin_amdgcn_cvt_pkrtz(x0.z, x0.w);
  u.h[2] = __builtin_amdgcn_cvt_pkrtz(x1.x, x1.y);
  u.h[3] = __builtin_amdgcn_cvt_pkrtz(x1.z, x1.w);
  u.h[4] = __builtin_amdgcn_cvt_pkrtz(x2.x, x2.y);
  u.h[5] = __builtin_amdgcn_cvt_pkrtz(x2.z, x2.w);
  u.h[6] = __builtin_amdgcn_cvt_pkrtz(x3.x, x3.y);
  u.h[7] = __builtin_amdgcn_cvt_pkrtz(x3.z, x3.w);
  return u.v;
}

// B fragment: 32x16 f16 tile from row-major fp32 W[k*M + col].
// lane holds column N=lane&15; element j covers k = kb + (j>>3)*16 + kh*8 + (j&7).
// Caller clamps col; cols >= M only affect unstored D columns.
__device__ inline v16h load_b_frag(const float* __restrict__ W, int col, int M,
                                   int kb, int kh) {
  const float* p = W + (size_t)(kb + (kh << 3)) * M + col;
  float x[16];
#pragma unroll
  for (int j = 0; j < 16; ++j) x[j] = p[(size_t)((j & 7) + ((j >> 3) << 4)) * M];
  union { v16h v; fp16x2 h[8]; } u;
#pragma unroll
  for (int j = 0; j < 8; ++j)
    u.h[j] = __builtin_amdgcn_cvt_pkrtz(x[2 * j], x[2 * j + 1]);
  return u.v;
}

// ---------------------------------------------------------------- kernels

__global__ void fill_kernel(float* p, float v, int n) {
  int i = blockIdx.x * blockDim.x + threadIdx.x;
  if (i < n) p[i] = v;
}

__global__ void deg_scatter(const int* __restrict__ dst, float* __restrict__ nrm, int E) {
  int i = blockIdx.x * blockDim.x + threadIdx.x;
  if (i < E) atomicAdd(&nrm[dst[i]], 1.0f);
}

__global__ void rsqrt_ip(float* p, int n) {
  int i = blockIdx.x * blockDim.x + threadIdx.x;
  if (i < n) p[i] = rsqrtf(p[i]);
}

// Dual GEMM for one GCN layer: hpre = agg = (A@W)*norm ; R = A@Wr
// One wave per 16x16 output tile; K multiple of 32; shares A fragments.
__global__ void gcn_gemm_dual(const float* __restrict__ A, const float* __restrict__ W,
                              const float* __restrict__ Wr, const float* __restrict__ nrm,
                              float* __restrict__ hpre, float* __restrict__ agg,
                              float* __restrict__ R, int nrows, int K, int M) {
  int wave = threadIdx.x >> 5, lane = threadIdx.x & 31;
  int tiles_m = (M + 15) >> 4;
  int tiles_n = (nrows + 15) >> 4;
  int tile = blockIdx.x * (blockDim.x >> 5) + wave;
  if (tile >= tiles_m * tiles_n) return;
  int tr = tile / tiles_m, tc = tile % tiles_m;
  int r   = min(tr * 16 + (lane & 15), nrows - 1);
  int col = min(tc * 16 + (lane & 15), M - 1);
  int kh  = lane >> 4;
  v8f c = {}; v8f cr = {};
  for (int kb = 0; kb < K; kb += 32) {
    v16h a  = load_a_frag(A, r, K, kb, kh);
    v16h b0 = load_b_frag(W,  col, M, kb, kh);
    v16h b1 = load_b_frag(Wr, col, M, kb, kh);
    c  = __builtin_amdgcn_wmma_f32_16x16x32_f16(false, a, false, b0, (short)0, c,  false, false);
    cr = __builtin_amdgcn_wmma_f32_16x16x32_f16(false, a, false, b1, (short)0, cr, false, false);
  }
  int colg  = tc * 16 + (lane & 15);
  int row0  = tr * 16 + (kh << 3);
  if (tr * 16 + 16 <= nrows && tc * 16 + 16 <= M) {
    // full tile (uniform branch): unguarded stores, strength-reduced addresses
    size_t base = (size_t)row0 * M + colg;
#pragma unroll
    for (int i = 0; i < 8; ++i) {
      float t = c[i] * nrm[row0 + i];
      size_t o = base + (size_t)i * M;
      hpre[o] = t;
      agg [o] = t;
      R   [o] = cr[i];
    }
  } else {
#pragma unroll
    for (int i = 0; i < 8; ++i) {
      int row = row0 + i;
      if (row < nrows && colg < M) {
        float t = c[i] * nrm[row];
        size_t o = (size_t)row * M + colg;
        hpre[o] = t;
        agg [o] = t;
        R   [o] = cr[i];
      }
    }
  }
}

// Generic guarded GEMM: C = A@W (+ bias), one wave per 16x16 tile.
__global__ void wmma_gemm_bias(const float* __restrict__ A, const float* __restrict__ W,
                               const float* __restrict__ bias, float* __restrict__ C,
                               int nrows, int K, int M) {
  int wave = threadIdx.x >> 5, lane = threadIdx.x & 31;
  int tiles_m = (M + 15) >> 4;
  int tiles_n = (nrows + 15) >> 4;
  int tile = blockIdx.x * (blockDim.x >> 5) + wave;
  if (tile >= tiles_m * tiles_n) return;
  int tr = tile / tiles_m, tc = tile % tiles_m;
  int r   = min(tr * 16 + (lane & 15), nrows - 1);
  int col = min(tc * 16 + (lane & 15), M - 1);
  int kh  = lane >> 4;
  v8f c = {};
  for (int kb = 0; kb < K; kb += 32) {
    v16h a = load_a_frag(A, r, K, kb, kh);
    v16h b = load_b_frag(W, col, M, kb, kh);
    c = __builtin_amdgcn_wmma_f32_16x16x32_f16(false, a, false, b, (short)0, c, false, false);
  }
  int colg = tc * 16 + (lane & 15);
  int row0 = tr * 16 + (kh << 3);
  float bv = (bias != nullptr) ? bias[min(colg, M - 1)] : 0.0f;
  if (tr * 16 + 16 <= nrows && tc * 16 + 16 <= M) {
    size_t base = (size_t)row0 * M + colg;
#pragma unroll
    for (int i = 0; i < 8; ++i)
      C[base + (size_t)i * M] = c[i] + bv;
  } else {
#pragma unroll
    for (int i = 0; i < 8; ++i) {
      int row = row0 + i;
      if (row < nrows && colg < M)
        C[(size_t)row * M + colg] = c[i] + bv;
    }
  }
}

// agg[dst] += hpre[src] over edges (float4 chunks)
__global__ void edge_scatter(const float* __restrict__ hpre, const int* __restrict__ src,
                             const int* __restrict__ dst, float* __restrict__ agg,
                             int E, int F) {
  int c4 = F >> 2;
  int idx = blockIdx.x * blockDim.x + threadIdx.x;
  if (idx >= E * c4) return;
  int e = idx / c4, c = idx - e * c4;
  const float4 v = *(const float4*)(hpre + (size_t)src[e] * F + c * 4);
  float* p = agg + (size_t)dst[e] * F + c * 4;
  atomicAdd(p + 0, v.x); atomicAdd(p + 1, v.y);
  atomicAdd(p + 2, v.z); atomicAdd(p + 3, v.w);
}

// h = agg*norm + R + br ; LayerNorm ; ReLU   (one wave per row, F in {64,128})
__global__ void gcn_finalize(const float* __restrict__ agg, const float* __restrict__ R,
                             const float* __restrict__ br, const float* __restrict__ nrm,
                             const float* __restrict__ g, const float* __restrict__ b,
                             float* __restrict__ out, int nrows, int F) {
  int wave = threadIdx.x >> 5, lane = threadIdx.x & 31;
  int row = blockIdx.x * (blockDim.x >> 5) + wave;
  if (row >= nrows) return;
  float nm = nrm[row];
  int FP = F >> 5;
  float hv[4];
  float s = 0.f;
  for (int j = 0; j < FP; ++j) {
    int f = lane + (j << 5);
    float h = agg[(size_t)row * F + f] * nm + R[(size_t)row * F + f] + br[f];
    hv[j] = h; s += h;
  }
  float mu = wave_sum(s) / F;
  float vs = 0.f;
  for (int j = 0; j < FP; ++j) { float d = hv[j] - mu; vs += d * d; }
  float inv = rsqrtf(wave_sum(vs) / F + LN_EPS);
  for (int j = 0; j < FP; ++j) {
    int f = lane + (j << 5);
    float y = (hv[j] - mu) * inv * g[f] + b[f];
    out[(size_t)row * F + f] = fmaxf(y, 0.f);
  }
}

__global__ void pool_sum(const float* __restrict__ h, const int* __restrict__ gid,
                         float* __restrict__ hgs, float* __restrict__ cnt, int N, int F) {
  int idx = blockIdx.x * blockDim.x + threadIdx.x;
  if (idx >= N * F) return;
  int node = idx / F, f = idx - node * F;
  int g = gid[node];
  atomicAdd(&hgs[(size_t)g * F + f], h[idx]);
  if (f == 0) atomicAdd(&cnt[g], 1.0f);
}

__global__ void pool_div(const float* __restrict__ hgs, const float* __restrict__ cnt,
                         float* __restrict__ hg, int Bn, int F) {
  int idx = blockIdx.x * blockDim.x + threadIdx.x;
  if (idx >= Bn * F) return;
  int g = idx / F;
  hg[idx] = hgs[idx] / fmaxf(cnt[g], 1.0f);
}

__global__ void ppf_gate(const float* __restrict__ mu, const float* __restrict__ lv,
                         const float* __restrict__ al, const float* __restrict__ d3,
                         float* __restrict__ v3, int n) {
  int i = blockIdx.x * blockDim.x + threadIdx.x;
  if (i >= n) return;
  float l = fminf(fmaxf(lv[i], -8.0f), 8.0f);
  float var = __expf(l) + 1e-6f;
  float z = (d3[i] - mu[i]) / (sqrtf(var) + 1e-6f);
  float w = (1.0f / (1.0f + __expf(-al[i]))) * fminf(1.0f / var, 50.0f);
  v3[i] = w * z;
}

// LayerNorm (+optional ReLU), one wave per row, F multiple of 32
__global__ void ln_act(const float* __restrict__ in, const float* __restrict__ g,
                       const float* __restrict__ b, float* __restrict__ out,
                       int nrows, int F, int do_relu) {
  int wave = threadIdx.x >> 5, lane = threadIdx.x & 31;
  int row = blockIdx.x * (blockDim.x >> 5) + wave;
  if (row >= nrows) return;
  int FP = F >> 5;
  float hv[4];
  float s = 0.f;
  for (int j = 0; j < FP; ++j) { hv[j] = in[(size_t)row * F + lane + (j << 5)]; s += hv[j]; }
  float mu = wave_sum(s) / F;
  float vs = 0.f;
  for (int j = 0; j < FP; ++j) { float d = hv[j] - mu; vs += d * d; }
  float inv = rsqrtf(wave_sum(vs) / F + LN_EPS);
  for (int j = 0; j < FP; ++j) {
    int f = lane + (j << 5);
    float y = (hv[j] - mu) * inv * g[f] + b[f];
    out[(size_t)row * F + f] = do_relu ? fmaxf(y, 0.f) : y;
  }
}

// fuse = LN(U*V)   (F = RANK = 64)
__global__ void mul_ln(const float* __restrict__ U, const float* __restrict__ V,
                       const float* __restrict__ g, const float* __restrict__ b,
                       float* __restrict__ out, int nrows, int F) {
  int wave = threadIdx.x >> 5, lane = threadIdx.x & 31;
  int row = blockIdx.x * (blockDim.x >> 5) + wave;
  if (row >= nrows) return;
  int FP = F >> 5;
  float hv[4];
  float s = 0.f;
  for (int j = 0; j < FP; ++j) {
    size_t o = (size_t)row * F + lane + (j << 5);
    hv[j] = U[o] * V[o]; s += hv[j];
  }
  float mu = wave_sum(s) / F;
  float vs = 0.f;
  for (int j = 0; j < FP; ++j) { float d = hv[j] - mu; vs += d * d; }
  float inv = rsqrtf(wave_sum(vs) / F + LN_EPS);
  for (int j = 0; j < FP; ++j) {
    int f = lane + (j << 5);
    out[(size_t)row * F + f] = (hv[j] - mu) * inv * g[f] + b[f];
  }
}

// BatchNorm batch stats (biased variance), one thread per feature column
__global__ void bn_stats(const float* __restrict__ h, float* __restrict__ m,
                         float* __restrict__ v, int rows, int F) {
  int f = threadIdx.x;
  if (f >= F) return;
  float s = 0.f, ss = 0.f;
  for (int r = 0; r < rows; ++r) {
    float x = h[(size_t)r * F + f];
    s += x; ss += x * x;
  }
  float mu = s / rows;
  m[f] = mu;
  v[f] = ss / rows - mu * mu;
}

// out[row] = relu(BN(h1)) @ Wh2 + bh2   (wave-per-row dot product, F=128)
__global__ void bn_relu_out(const float* __restrict__ h, const float* __restrict__ m,
                            const float* __restrict__ v, const float* __restrict__ g,
                            const float* __restrict__ b, const float* __restrict__ W,
                            const float* __restrict__ bo, float* __restrict__ out,
                            int rows, int F) {
  int wave = threadIdx.x >> 5, lane = threadIdx.x & 31;
  int row = blockIdx.x * (blockDim.x >> 5) + wave;
  if (row >= rows) return;
  int FP = F >> 5;
  float acc = 0.f;
  for (int j = 0; j < FP; ++j) {
    int f = lane + (j << 5);
    float hn = (h[(size_t)row * F + f] - m[f]) * rsqrtf(v[f] + LN_EPS) * g[f] + b[f];
    acc += fmaxf(hn, 0.f) * W[f];
  }
  acc = wave_sum(acc);
  if (lane == 0) out[row] = acc + bo[0];
}

// ---------------------------------------------------------------- launcher

extern "C" void kernel_launch(void* const* d_in, const int* in_sizes, int n_in,
                              void* d_out, int out_size, void* d_ws, size_t ws_size,
                              hipStream_t stream) {
  (void)n_in; (void)out_size; (void)ws_size;

  const float* x     = (const float*)d_in[0];
  const float* desc3 = (const float*)d_in[2];
  const int*   src   = (const int*)d_in[3];
  const int*   dst   = (const int*)d_in[4];
  const int*   gid   = (const int*)d_in[5];
  const float* W1  = (const float*)d_in[6];
  const float* W1r = (const float*)d_in[7];
  const float* b1r = (const float*)d_in[8];
  const float* ln1g = (const float*)d_in[9];
  const float* ln1b = (const float*)d_in[10];
  const float* W2  = (const float*)d_in[11];
  const float* W2r = (const float*)d_in[12];
  const float* b2r = (const float*)d_in[13];
  const float* ln2g = (const float*)d_in[14];
  const float* ln2b = (const float*)d_in[15];
  const float* Wmu = (const float*)d_in[16];
  const float* bmu = (const float*)d_in[17];
  const float* Wlv = (const float*)d_in[18];
  const float* blv = (const float*)d_in[19];
  const float* Wa  = (const float*)d_in[20];
  const float* ba  = (const float*)d_in[21];
  const float* Wvr = (const float*)d_in[22];
  const float* bvr = (const float*)d_in[23];
  const float* lnvg = (const float*)d_in[24];
  const float* lnvb = (const float*)d_in[25];
  const float* WU  = (const float*)d_in[26];
  const float* WV  = (const float*)d_in[27];
  const float* lnfg = (const float*)d_in[28];
  const float* lnfb = (const float*)d_in[29];
  const float* Wh1 = (const float*)d_in[30];
  const float* bh1 = (const float*)d_in[31];
  const float* bng = (const float*)d_in[32];
  const float* bnb = (const float*)d_in[33];
  const float* Wh2 = (const float*)d_in[34];
  const float* bh2 = (const float*)d_in[35];

  const int N  = in_sizes[0] / 64;
  const int E  = in_sizes[3];
  const int Bn = in_sizes[2] / 256;

  float* wsf = (float*)d_ws;
  size_t off = 0;
  auto alloc = [&](size_t n) { float* p = wsf + off; off += (n + 63) & ~(size_t)63; return p; };

  float* nrm  = alloc(N);
  float* hpre = alloc((size_t)N * 128);   // reused layer 2 (N*64)
  float* agg  = alloc((size_t)N * 128);   // reused layer 2
  float* Rb   = alloc((size_t)N * 128);   // reused layer 2
  float* h1n  = alloc((size_t)N * 128);
  float* h2n  = alloc((size_t)N * 64);
  float* hgs  = alloc((size_t)Bn * 64);
  float* cntb = alloc(Bn);
  float* hg   = alloc((size_t)Bn * 64);
  float* mu   = alloc((size_t)Bn * 256);
  float* lv   = alloc((size_t)Bn * 256);
  float* al   = alloc((size_t)Bn * 256);
  float* v3   = alloc((size_t)Bn * 256);
  float* vrp  = alloc((size_t)Bn * 32);
  float* vr   = alloc((size_t)Bn * 32);
  float* Ub   = alloc((size_t)Bn * 64);
  float* Vb   = alloc((size_t)Bn * 64);
  float* fuse = alloc((size_t)Bn * 64);
  float* h1b  = alloc((size_t)Bn * 128);
  float* bnm  = alloc(128);
  float* bnv  = alloc(128);

  // degree -> norm = (deg+1)^-0.5 (self-loop via init 1.0); zero pool accumulators
  fill_kernel<<<(N + 255) / 256, 256, 0, stream>>>(nrm, 1.0f, N);
  fill_kernel<<<(Bn * 64 + 255) / 256, 256, 0, stream>>>(hgs, 0.0f, Bn * 64);
  fill_kernel<<<(Bn + 255) / 256, 256, 0, stream>>>(cntb, 0.0f, Bn);
  deg_scatter<<<(E + 255) / 256, 256, 0, stream>>>(dst, nrm, E);
  rsqrt_ip<<<(N + 255) / 256, 256, 0, stream>>>(nrm, N);

  auto gemm_dual = [&](const float* A, const float* Wm, const float* Wrm,
                       float* hp, float* ag, float* Rr, int nr, int K, int M) {
    int tiles = ((M + 15) / 16) * ((nr + 15) / 16);
    gcn_gemm_dual<<<(tiles + 7) / 8, 256, 0, stream>>>(A, Wm, Wrm, nrm, hp, ag, Rr, nr, K, M);
  };
  auto gemm = [&](const float* A, const float* Wm, const float* bias,
                  float* C, int nr, int K, int M) {
    int tiles = ((M + 15) / 16) * ((nr + 15) / 16);
    wmma_gemm_bias<<<(tiles + 7) / 8, 256, 0, stream>>>(A, Wm, bias, C, nr, K, M);
  };

  // ---- GCN layer 1 (64 -> 128)
  gemm_dual(x, W1, W1r, hpre, agg, Rb, N, 64, 128);
  edge_scatter<<<(E * 32 + 255) / 256, 256, 0, stream>>>(hpre, src, dst, agg, E, 128);
  gcn_finalize<<<(N + 7) / 8, 256, 0, stream>>>(agg, Rb, b1r, nrm, ln1g, ln1b, h1n, N, 128);

  // ---- GCN layer 2 (128 -> 64)
  gemm_dual(h1n, W2, W2r, hpre, agg, Rb, N, 128, 64);
  edge_scatter<<<(E * 16 + 255) / 256, 256, 0, stream>>>(hpre, src, dst, agg, E, 64);
  gcn_finalize<<<(N + 7) / 8, 256, 0, stream>>>(agg, Rb, b2r, nrm, ln2g, ln2b, h2n, N, 64);

  // ---- per-graph mean pool
  pool_sum<<<(N * 64 + 255) / 256, 256, 0, stream>>>(h2n, gid, hgs, cntb, N, 64);
  pool_div<<<(Bn * 64 + 255) / 256, 256, 0, stream>>>(hgs, cntb, hg, Bn, 64);

  // ---- PPF gaussian gate
  gemm(hg, Wmu, bmu, mu, Bn, 64, 256);
  gemm(hg, Wlv, blv, lv, Bn, 64, 256);
  gemm(hg, Wa,  ba,  al, Bn, 64, 256);
  ppf_gate<<<(Bn * 256 + 255) / 256, 256, 0, stream>>>(mu, lv, al, desc3, v3, Bn * 256);

  // ---- v reduction + low-rank bilinear fuse
  gemm(v3, Wvr, bvr, vrp, Bn, 256, 32);
  ln_act<<<(Bn + 7) / 8, 256, 0, stream>>>(vrp, lnvg, lnvb, vr, Bn, 32, 1);
  gemm(hg, WU, nullptr, Ub, Bn, 64, 64);
  gemm(vr, WV, nullptr, Vb, Bn, 32, 64);
  mul_ln<<<(Bn + 7) / 8, 256, 0, stream>>>(Ub, Vb, lnfg, lnfb, fuse, Bn, 64);

  // ---- head MLP: Linear -> BatchNorm(batch stats) -> ReLU -> Linear
  gemm(fuse, Wh1, bh1, h1b, Bn, 64, 128);
  bn_stats<<<1, 128, 0, stream>>>(h1b, bnm, bnv, Bn, 128);
  bn_relu_out<<<(Bn + 7) / 8, 256, 0, stream>>>(h1b, bnm, bnv, bng, bnb, Wh2, bh2,
                                                (float*)d_out, Bn, 128);
}
